// YuKeMLPFLOW_onlyX_37460704756280
// MI455X (gfx1250) — matrix operand order repaired and checked by reading, not compiled, not run on hardware
//
#include <hip/hip_runtime.h>
#include <hip/hip_bf16.h>
#include <math.h>

// ---------------------------------------------------------------------------
// YuKe MLP-flow: fwd 12->64->64->64->12 (LeakyReLU 0.2) + last Jacobian row.
// CDNA5 (gfx1250) wave32 WMMA f16 implementation.
//   - activations ride the B operand (batch=N=16 per wave-tile)
//   - weights ride the A operand, pre-packed to VGPR layout, staged in LDS
//     (lo/hi split, 16B lane stride -> bank-conflict-free ds_load_b128)
//   - opaque per-iteration LDS offsets defeat LICM so fragments are re-read
//     from LDS each tile instead of being promoted to (spilled) registers
//   - D->B relayout via ds_swizzle XOR-16 butterfly + f16 packing
// ---------------------------------------------------------------------------

typedef _Float16 v16h __attribute__((ext_vector_type(16)));
typedef float    v8f  __attribute__((ext_vector_type(8)));
typedef _Float16 h2   __attribute__((ext_vector_type(2)));

#define SLOPE 0.2f
#define DD 12
#define HH 64
#define NFRAG 40
#define WS_FRAG_BYTES (NFRAG * 1024)   // 40 fragments x 32 lanes x 32B

union Frag { v16h h; unsigned u[8]; float4 q[2]; };

static __device__ __forceinline__ unsigned pk2f(float a, float b) {
  h2 v; v.x = (_Float16)a; v.y = (_Float16)b;
  return __builtin_bit_cast(unsigned, v);
}
static __device__ __forceinline__ unsigned pkmul(unsigned a, unsigned b) {
  h2 x = __builtin_bit_cast(h2, a), y = __builtin_bit_cast(h2, b);
  h2 r = x * y;                       // v_pk_mul_f16
  return __builtin_bit_cast(unsigned, r);
}
static __device__ __forceinline__ v8f wmma16(v16h a, v16h b, v8f c) {
  // D = A(16x32 f16) * B(32x16 f16) + C(16x16 f32)
  return __builtin_amdgcn_wmma_f32_16x16x32_f16(false, a, false, b,
                                                (short)0, c, false, false);
}

// D-layout packed-f16 (4 feature-tiles x 4 dwords per lane) -> two B fragments
// (K-chunks 0/1). Lane pair (b,hi)<->(b,hi^1) exchanges via ds_swizzle SWAPX16.
static __device__ __forceinline__ void dtob(const unsigned (&pk)[4][4], int hi,
                                            Frag& f0, Frag& f1) {
  unsigned sw[4][4];
#pragma unroll
  for (int t = 0; t < 4; ++t)
#pragma unroll
    for (int d = 0; d < 4; ++d)
      sw[t][d] = (unsigned)__builtin_amdgcn_ds_swizzle((int)pk[t][d], 0x401F);
#pragma unroll
  for (int kc = 0; kc < 2; ++kc) {
    Frag& f = kc ? f1 : f0;
#pragma unroll
    for (int d = 0; d < 4; ++d) {
      f.u[d]     = hi ? sw[2*kc+1][d] : pk[2*kc][d];
      f.u[4 + d] = hi ? pk[2*kc+1][d] : sw[2*kc][d];
    }
  }
}

// ---------------------------------------------------------------------------
// Pack kernel: build all 40 A-operand fragments (f16, exact VGPR layout,
// lo/hi split) and the bias / W4-last-row side data in d_ws.
// Fragment map: 0-3 W1fwd | 4-11 W2fwd | 12-19 W3fwd | 20-21 W4fwd
//               22-29 W3T  | 30-37 W2T  | 38-39 W1T
// ---------------------------------------------------------------------------
__global__ void yuke_pack_kernel(const float* __restrict__ W1, const float* __restrict__ W2,
                                 const float* __restrict__ W3, const float* __restrict__ W4,
                                 const float* __restrict__ b1, const float* __restrict__ b2,
                                 const float* __restrict__ b3, const float* __restrict__ b4,
                                 _Float16* __restrict__ frag, float* __restrict__ biasArea) {
  int g = blockIdx.x * blockDim.x + threadIdx.x;
  if (g < NFRAG * 32) {
    int fi = g >> 5, lane = g & 31;
    int m = lane & 15, hi = lane >> 4;
    _Float16 vals[16];
#pragma unroll
    for (int j = 0; j < 16; ++j) {
      // A fragment in this code's contiguous k order (B matches): half j of lane -> K = 16*hi + j
      int k = 16 * hi + j;
      float v = 0.f;
      if (fi < 4)       { int t = fi;                            if (k < DD) v = W1[(16*t + m) * DD + k]; }
      else if (fi < 12) { int q = fi - 4,  t = q >> 1, kc = q&1; v = W2[(16*t + m) * HH + 32*kc + k]; }
      else if (fi < 20) { int q = fi - 12, t = q >> 1, kc = q&1; v = W3[(16*t + m) * HH + 32*kc + k]; }
      else if (fi < 22) { int kc = fi - 20;                      if (m < DD) v = W4[m * HH + 32*kc + k]; }
      else if (fi < 30) { int q = fi - 22, t = q >> 1, kc = q&1; v = W3[(32*kc + k) * HH + 16*t + m]; }
      else if (fi < 38) { int q = fi - 30, t = q >> 1, kc = q&1; v = W2[(32*kc + k) * HH + 16*t + m]; }
      else              { int kc = fi - 38;                      if (m < DD) v = W1[(32*kc + k) * DD + m]; }
      vals[j] = (_Float16)v;
    }
    // lo region: halves 0-7, hi region: halves 8-15 (16B per lane each)
    _Float16* dlo = frag + (size_t)g * 8;
    _Float16* dhi = frag + (size_t)NFRAG * 32 * 8 + (size_t)g * 8;
#pragma unroll
    for (int j = 0; j < 8; ++j) { dlo[j] = vals[j]; dhi[j] = vals[8 + j]; }
  }
  // Bias block: f32, D-layout addressing [hi][layer][t][r] = b_l[16t+8hi+r]
  if (g < 192) {
    int hi = g / 96, rem = g % 96, l = rem / 32, rr = rem % 32, t = rr / 8, r = rr % 8;
    const float* bl = (l == 0) ? b1 : (l == 1) ? b2 : b3;
    biasArea[g] = bl[16*t + 8*hi + r];
  }
  if (g < 16) { int hi = g / 8, r = g % 8, n = 8*hi + r; biasArea[192 + g] = (n < DD) ? b4[n] : 0.f; }
  if (g < 32) {                      // packed-f16 W4 last row, D-aligned
    int hi = g / 16, rem = g % 16, t = rem / 4, d = rem % 4;
    int f0 = 16*t + 8*hi + 2*d;
    ((unsigned*)(biasArea + 208))[g] = pk2f(W4[11*HH + f0], W4[11*HH + f0 + 1]);
  }
}

// ---------------------------------------------------------------------------
// Main kernel: one wave = one 16-sample tile per iteration (grid-stride).
// ---------------------------------------------------------------------------
__global__ void __launch_bounds__(256)
yuke_mlp_flow_kernel(const float* __restrict__ x, const void* __restrict__ ws,
                     float* __restrict__ out, int nTiles) {
  const int lane = threadIdx.x & 31;
  const int wave = threadIdx.x >> 5;
  const int b    = lane & 15;
  const int hi   = lane >> 4;

  __shared__ __align__(16) _Float16 sFrag[NFRAG * 32 * 16];  // 40KB lo|hi
  __shared__ __align__(16) float    sBias[192];
  __shared__ __align__(16) float    sB4[16];
  __shared__ unsigned               sW4L[32];
  {
    // stage fragment table (40KB) + side data into LDS
    const float4* src = (const float4*)ws;
    float4* dst = (float4*)sFrag;
#pragma unroll
    for (int i = 0; i < 10; ++i)
      dst[threadIdx.x + 256 * i] = src[threadIdx.x + 256 * i];
    const float* wsb = (const float*)((const char*)ws + WS_FRAG_BYTES);
    int t = threadIdx.x;
    if (t < 192) sBias[t] = wsb[t];
    if (t < 16)  sB4[t]   = wsb[192 + t];
    if (t < 32)  sW4L[t]  = ((const unsigned*)(wsb + 208))[t];
  }
  __syncthreads();

  // wave-constant packed W4 last row -> registers (intentional hoist, 16 regs)
  unsigned w4l[16];
#pragma unroll
  for (int i = 0; i < 16; ++i) w4l[i] = sW4L[hi * 16 + i];

  const float4* fragLo = (const float4*)sFrag;           // [NFRAG*32] 16B chunks
  const float4* fragHi = fragLo + NFRAG * 32;

#define LOAD_A(fr, fi)                                                         \
  { (fr).q[0] = fragLo[(fi) * 32 + lane + fo];                                 \
    (fr).q[1] = fragHi[(fi) * 32 + lane + fo]; }

#define LOAD_BIAS(cv, lidx, t)                                                 \
  { const float4* bp = (const float4*)&sBias[hi * 96 + (lidx) * 32 + (t) * 8]; \
    float4 c0 = bp[bo], c1 = bp[bo + 1];                                       \
    (cv)[0]=c0.x; (cv)[1]=c0.y; (cv)[2]=c0.z; (cv)[3]=c0.w;                    \
    (cv)[4]=c1.x; (cv)[5]=c1.y; (cv)[6]=c1.z; (cv)[7]=c1.w; }

  const int wavesTotal = gridDim.x * (blockDim.x >> 5);
#pragma clang loop unroll(disable)
  for (int tile = blockIdx.x * (blockDim.x >> 5) + wave; tile < nTiles;
       tile += wavesTotal) {
    // Opaque zero offsets, redefined every iteration: LDS loads below cannot
    // be hoisted out of the loop (they would otherwise be register-promoted
    // and spilled to scratch).
    unsigned fo = 0, bo = 0;
    asm volatile("" : "+v"(fo), "+v"(bo));

    const size_t row = (size_t)tile * 16 + b;
    __builtin_prefetch((const void*)(x + ((size_t)(tile + wavesTotal) * 16 + b) * DD), 0, 1);

    // ---- B fragment for x (K-chunk 0 only; K=12..31 zero-padded) ----------
    Frag bf0, bf1;
#pragma unroll
    for (int d = 0; d < 8; ++d) bf0.u[d] = 0u;
    if (hi == 0) {
      const float4* xp = (const float4*)(x + row * DD);
      float4 p0 = xp[0], p1 = xp[1], p2 = xp[2];
      bf0.u[0] = pk2f(p0.x, p0.y); bf0.u[1] = pk2f(p0.z, p0.w);
      bf0.u[2] = pk2f(p1.x, p1.y); bf0.u[3] = pk2f(p1.z, p1.w);
      bf0.u[4] = pk2f(p2.x, p2.y); bf0.u[5] = pk2f(p2.z, p2.w);
    }

    v8f acc[4];
    unsigned hpk[4][4], dpk1[4][4], dpk2[4][4], dpk3[4][4];

    // ---- layer 1: z1^T = W1 * x^T + b1 (single K-chunk) -------------------
#pragma unroll
    for (int t = 0; t < 4; ++t) {
      v8f c; LOAD_BIAS(c, 0, t);
      Frag a; LOAD_A(a, t);
      acc[t] = wmma16(a.h, bf0.h, c);
    }
#pragma unroll
    for (int t = 0; t < 4; ++t)
#pragma unroll
      for (int d = 0; d < 4; ++d) {
        float z0 = acc[t][2*d], z1 = acc[t][2*d + 1];
        float g0 = z0 > 0.f ? 1.f : SLOPE, g1 = z1 > 0.f ? 1.f : SLOPE;
        hpk[t][d]  = pk2f(z0 * g0, z1 * g1);
        dpk1[t][d] = pk2f(g0, g1);
      }
    dtob(hpk, hi, bf0, bf1);

    // ---- layers 2,3 -------------------------------------------------------
#define DENSE_LAYER(base, lidx, dpkArr)                                        \
    { _Pragma("unroll") for (int t = 0; t < 4; ++t) {                          \
        v8f c; LOAD_BIAS(c, lidx, t);                                          \
        Frag a0; LOAD_A(a0, (base) + 2*t);                                     \
        Frag a1; LOAD_A(a1, (base) + 2*t + 1);                                 \
        c = wmma16(a0.h, bf0.h, c);                                            \
        c = wmma16(a1.h, bf1.h, c);                                            \
        acc[t] = c; }                                                          \
      _Pragma("unroll") for (int t = 0; t < 4; ++t)                            \
        _Pragma("unroll") for (int d = 0; d < 4; ++d) {                        \
          float z0 = acc[t][2*d], z1 = acc[t][2*d + 1];                        \
          float g0 = z0 > 0.f ? 1.f : SLOPE, g1 = z1 > 0.f ? 1.f : SLOPE;      \
          hpk[t][d] = pk2f(z0 * g0, z1 * g1);                                  \
          dpkArr[t][d] = pk2f(g0, g1); }                                       \
      dtob(hpk, hi, bf0, bf1); }

    DENSE_LAYER(4, 1, dpk2)
    DENSE_LAYER(12, 2, dpk3)

    // ---- layer 4: residual^T = W4 * h3^T + b4 -----------------------------
    v8f c4;
    {
      const float4* bp = (const float4*)&sB4[hi * 8];
      float4 c0 = bp[bo], c1 = bp[bo + 1];
      c4[0]=c0.x; c4[1]=c0.y; c4[2]=c0.z; c4[3]=c0.w;
      c4[4]=c1.x; c4[5]=c1.y; c4[6]=c1.z; c4[7]=c1.w;
    }
    { Frag a0; LOAD_A(a0, 20); Frag a1; LOAD_A(a1, 21);
      c4 = wmma16(a0.h, bf0.h, c4);
      c4 = wmma16(a1.h, bf1.h, c4); }
    {
      float* op = out + row * DD + hi * 8;
      if (hi == 0) {
        ((float4*)op)[0] = make_float4(c4[0], c4[1], c4[2], c4[3]);
        ((float4*)op)[1] = make_float4(c4[4], c4[5], c4[6], c4[7]);
      } else {
        ((float4*)op)[0] = make_float4(c4[0], c4[1], c4[2], c4[3]);
      }
    }

    // ---- backward: u0 = W4[-1] * d3 ---------------------------------------
#pragma unroll
    for (int t = 0; t < 4; ++t)
#pragma unroll
      for (int d = 0; d < 4; ++d)
        hpk[t][d] = pkmul(w4l[t * 4 + d], dpk3[t][d]);
    dtob(hpk, hi, bf0, bf1);

#define BWD_LAYER(base, dpkArr)                                                \
    { _Pragma("unroll") for (int t = 0; t < 4; ++t) {                          \
        v8f c = {};                                                            \
        Frag a0; LOAD_A(a0, (base) + 2*t);                                     \
        Frag a1; LOAD_A(a1, (base) + 2*t + 1);                                 \
        c = wmma16(a0.h, bf0.h, c);                                            \
        c = wmma16(a1.h, bf1.h, c);                                            \
        acc[t] = c; }                                                          \
      _Pragma("unroll") for (int t = 0; t < 4; ++t)                            \
        _Pragma("unroll") for (int d = 0; d < 4; ++d)                          \
          hpk[t][d] = pkmul(pk2f(acc[t][2*d], acc[t][2*d + 1]), dpkArr[t][d]); \
      dtob(hpk, hi, bf0, bf1); }

    BWD_LAYER(22, dpk2)   // u = (u @ W3) * d2
    BWD_LAYER(30, dpk1)   // u = (u @ W2) * d1

    // last_row^T = W1^T * u^T   (single 16-wide output tile)
    v8f cl = {};
    { Frag a0; LOAD_A(a0, 38); Frag a1; LOAD_A(a1, 39);
      cl = wmma16(a0.h, bf0.h, cl);
      cl = wmma16(a1.h, bf1.h, cl); }
    {
      float* op = out + (size_t)nTiles * 192 + row * DD + hi * 8;
      if (hi == 0) {
        ((float4*)op)[0] = make_float4(__logf(fabsf(cl[0])), __logf(fabsf(cl[1])),
                                       __logf(fabsf(cl[2])), __logf(fabsf(cl[3])));
        ((float4*)op)[1] = make_float4(__logf(fabsf(cl[4])), __logf(fabsf(cl[5])),
                                       __logf(fabsf(cl[6])), __logf(fabsf(cl[7])));
      } else {
        ((float4*)op)[0] = make_float4(__logf(fabsf(cl[0])), __logf(fabsf(cl[1])),
                                       __logf(fabsf(cl[2])), __logf(fabsf(cl[3])));
      }
    }
#undef DENSE_LAYER
#undef BWD_LAYER
  }
#undef LOAD_A
#undef LOAD_BIAS
}

extern "C" void kernel_launch(void* const* d_in, const int* in_sizes, int n_in,
                              void* d_out, int out_size, void* d_ws, size_t ws_size,
                              hipStream_t stream) {
  const float* x  = (const float*)d_in[0];
  const float* W1 = (const float*)d_in[1];
  const float* b1 = (const float*)d_in[2];
  const float* W2 = (const float*)d_in[3];
  const float* b2 = (const float*)d_in[4];
  const float* W3 = (const float*)d_in[5];
  const float* b3 = (const float*)d_in[6];
  const float* W4 = (const float*)d_in[7];
  const float* b4 = (const float*)d_in[8];
  float* out = (float*)d_out;

  const int Bn     = in_sizes[0] / DD;   // 524288
  const int nTiles = Bn / 16;            // 32768

  _Float16* frag  = (_Float16*)d_ws;
  float* biasArea = (float*)((char*)d_ws + WS_FRAG_BYTES);

  yuke_pack_kernel<<<dim3((NFRAG * 32 + 255) / 256), dim3(256), 0, stream>>>(
      W1, W2, W3, W4, b1, b2, b3, b4, frag, biasArea);

  yuke_mlp_flow_kernel<<<dim3(512), dim3(256), 0, stream>>>(x, d_ws, out, nTiles);
}
